// EnhancedTransformerBlock_51917564674691
// MI455X (gfx1250) — matrix operand order repaired and checked by reading, not compiled, not run on hardware
//
#include <hip/hip_runtime.h>

typedef __attribute__((ext_vector_type(16))) _Float16 v16h;
typedef __attribute__((ext_vector_type(8)))  _Float16 v8h;
typedef __attribute__((ext_vector_type(8)))  float    v8f;

#define DEVINL __device__ __forceinline__

static constexpr int BATCH = 4;
static constexpr int SEQ   = 2048;
static constexpr int DIM   = 256;
static constexpr int TOK   = BATCH * SEQ;   // 8192

union FragU {
  v16h v;
  struct { v8h lo, hi; } s;
};

DEVINL v8f zero8() { v8f z; for (int i = 0; i < 8; ++i) z[i] = 0.f; return z; }

DEVINL float fast_rcp(float x) { return __builtin_amdgcn_rcpf(x); }

// A-matrix 16x32 f16 fragment (M rows = lane&15).
// lane<16: K {0..7, 16..23}; lane>=16: K {8..15, 24..31}
DEVINL v16h load_frag_a(const _Float16* base, int ld) {
  const int lane = threadIdx.x & 31;
  const _Float16* p = base + (size_t)(lane & 15) * ld + ((lane & 16) ? 8 : 0);
  FragU u;
  u.s.lo = *reinterpret_cast<const v8h*>(p);
  u.s.hi = *reinterpret_cast<const v8h*>(p + 16);
  return u.v;
}

// B-matrix 32x16 f16 fragment (N cols = lane&15), per-lane K:
// lane<16: K {0..7, 16..23}; lane>=16: K {8..15, 24..31}. base points at row n0 of the
// K-contiguous operand (weights [N,K] or transposed V [hd,S]).
DEVINL v16h load_frag_b(const _Float16* base, int ld) {
  const int lane = threadIdx.x & 31;
  const _Float16* p = base + (size_t)(lane & 15) * ld + ((lane & 16) ? 8 : 0);
  FragU u;
  u.s.lo = *reinterpret_cast<const v8h*>(p);
  u.s.hi = *reinterpret_cast<const v8h*>(p + 16);
  return u.v;
}

DEVINL v8f wmma32(v16h a, v16h b, v8f c) {
  return __builtin_amdgcn_wmma_f32_16x16x32_f16(false, a, false, b, (short)0, c,
                                                false, false);
}

// ---------------------------------------------------------------- f32 -> f16
__global__ void cvt_f32_f16_kernel(const float* __restrict__ in,
                                   _Float16* __restrict__ out, int n) {
  int i = blockIdx.x * blockDim.x + threadIdx.x;
  if (i < n) out[i] = (_Float16)in[i];
}

// ------------------------------------------------- generic WMMA GEMM
// Y[M,N] = act(A[M,K] @ W[N,K]^T + bias), ACT: 0=none, 1=silu.
// Block tile 64x128 (4 waves), wave tile 32x64 (2x4 WMMA frags).
// HASF: write f32 to Cf; HASH: write f16 to Ch (leading dim ldc).
template <int ACT, bool HASF, bool HASH>
__global__ __launch_bounds__(128) void gemm_f16_kernel(
    const _Float16* __restrict__ A, const _Float16* __restrict__ W,
    const float* __restrict__ bias, float* __restrict__ Cf,
    _Float16* __restrict__ Ch, int K, int ldc) {
  const int lane = threadIdx.x & 31;
  const int wave = threadIdx.x >> 5;
  const int m0 = blockIdx.y * 64 + (wave >> 1) * 32;
  const int n0 = blockIdx.x * 128 + (wave & 1) * 64;

  v8f c[2][4];
#pragma unroll
  for (int i = 0; i < 2; ++i)
#pragma unroll
    for (int j = 0; j < 4; ++j) c[i][j] = zero8();

  for (int k = 0; k < K; k += 32) {
    v16h a[2], b[4];
#pragma unroll
    for (int i = 0; i < 2; ++i)
      a[i] = load_frag_a(A + (size_t)(m0 + 16 * i) * K + k, K);
#pragma unroll
    for (int j = 0; j < 4; ++j)
      b[j] = load_frag_b(W + (size_t)(n0 + 16 * j) * K + k, K);
#pragma unroll
    for (int i = 0; i < 2; ++i)
#pragma unroll
      for (int j = 0; j < 4; ++j) c[i][j] = wmma32(a[i], b[j], c[i][j]);
  }

  const int crow = (lane & 16) ? 8 : 0;
  const int ccol = lane & 15;

  float bv[4];
#pragma unroll
  for (int j = 0; j < 4; ++j) bv[j] = bias[n0 + 16 * j + ccol];

#pragma unroll
  for (int i = 0; i < 2; ++i) {
#pragma unroll
    for (int r = 0; r < 8; ++r) {
      const size_t rowoff =
          (size_t)(m0 + i * 16 + crow + r) * ldc + n0 + ccol;
#pragma unroll
      for (int j = 0; j < 4; ++j) {
        float v = c[i][j][r] + bv[j];
        if (ACT == 1) v = v * fast_rcp(1.f + __expf(-v));  // silu
        if (HASF) Cf[rowoff + 16 * j] = v;
        if (HASH) Ch[rowoff + 16 * j] = (_Float16)v;
      }
    }
  }
}

// -------------------------------------- V transpose for global attention
// vt[(b*8+h)*32 + d][s] = qkv[(b*S+s)*768 + 512 + h*32 + d]
__global__ __launch_bounds__(256) void transpose_vg_kernel(
    const _Float16* __restrict__ qkv, _Float16* __restrict__ vt) {
  int t = blockIdx.x;          // token 0..8191
  int b = t >> 11, s = t & (SEQ - 1);
  int tid = threadIdx.x;       // h*32 + d
  vt[((size_t)(b * 8) * 32 + tid) * SEQ + s] = qkv[(size_t)t * 768 + 512 + tid];
}

// -------------------------------------- global attention (flash, per wave)
// One wave owns a 16-row Q tile of one (batch, head). hd = 32 == WMMA K.
// 64 keys per iteration: 4 score WMMA + 4 PV WMMA, one shfl-reduction pass.
__global__ __launch_bounds__(128) void flash_global_kernel(
    const _Float16* __restrict__ qkv, const _Float16* __restrict__ vt,
    _Float16* __restrict__ out) {
  const int lane = threadIdx.x & 31;
  const int wave = threadIdx.x >> 5;
  const int b = blockIdx.z, h = blockIdx.y;
  const int q0 = blockIdx.x * 64 + wave * 16;

  const _Float16* qbase = qkv + (size_t)b * SEQ * 768 + h * 32;
  const _Float16* kbase = qbase + 256;
  const _Float16* vtb = vt + (size_t)(b * 8 + h) * 32 * SEQ;

  __shared__ __align__(16) _Float16 plds[4][16 * 64];
  _Float16* pl = plds[wave];

  v16h qf = load_frag_a(qbase + (size_t)q0 * 768, 768);
  const _Float16 qs = (_Float16)0.17677669529663687f;  // 1/sqrt(32)
#pragma unroll
  for (int i = 0; i < 16; ++i) qf[i] = qf[i] * qs;

  v8f o0 = zero8(), o1 = zero8();
  float mrow[8], lrow[8];
#pragma unroll
  for (int r = 0; r < 8; ++r) { mrow[r] = -1e30f; lrow[r] = 0.f; }
  const int crow = (lane & 16) ? 8 : 0;
  const int ccol = lane & 15;

  for (int j = 0; j < SEQ; j += 64) {
    v8f s[4];
#pragma unroll
    for (int t = 0; t < 4; ++t) {
      v16h kb = load_frag_b(kbase + (size_t)(j + 16 * t) * 768, 768);
      s[t] = wmma32(qf, kb, zero8());
    }

#pragma unroll
    for (int r = 0; r < 8; ++r) {
      float mx = fmaxf(fmaxf(s[0][r], s[1][r]), fmaxf(s[2][r], s[3][r]));
      mx = fmaxf(mx, __shfl_xor(mx, 1));
      mx = fmaxf(mx, __shfl_xor(mx, 2));
      mx = fmaxf(mx, __shfl_xor(mx, 4));
      mx = fmaxf(mx, __shfl_xor(mx, 8));
      float mnew = fmaxf(mrow[r], mx);
      float sc = __expf(mrow[r] - mnew);
      mrow[r] = mnew;
      float rs = 0.f;
#pragma unroll
      for (int t = 0; t < 4; ++t) {
        float p = __expf(s[t][r] - mnew);
        rs += p;
        pl[(crow + r) * 64 + 16 * t + ccol] = (_Float16)p;
      }
      rs += __shfl_xor(rs, 1);
      rs += __shfl_xor(rs, 2);
      rs += __shfl_xor(rs, 4);
      rs += __shfl_xor(rs, 8);
      lrow[r] = lrow[r] * sc + rs;
      o0[r] *= sc;
      o1[r] *= sc;
    }
    __syncthreads();
    v16h pf0 = load_frag_a(pl, 64);        // P keys j..j+31
    v16h pf1 = load_frag_a(pl + 32, 64);   // P keys j+32..j+63
    __syncthreads();

    v16h vb00 = load_frag_b(vtb + j, SEQ);                          // hd 0..15
    v16h vb01 = load_frag_b(vtb + (size_t)16 * SEQ + j, SEQ);       // hd 16..31
    v16h vb10 = load_frag_b(vtb + j + 32, SEQ);
    v16h vb11 = load_frag_b(vtb + (size_t)16 * SEQ + j + 32, SEQ);
    o0 = wmma32(pf0, vb00, o0);
    o0 = wmma32(pf1, vb10, o0);
    o1 = wmma32(pf0, vb01, o1);
    o1 = wmma32(pf1, vb11, o1);
  }

#pragma unroll
  for (int r = 0; r < 8; ++r) {
    int row = q0 + crow + r;
    float inv = fast_rcp(lrow[r]);
    size_t base = ((size_t)b * SEQ + row) * DIM + h * 32;
    out[base + ccol] = (_Float16)(o0[r] * inv);
    out[base + 16 + ccol] = (_Float16)(o1[r] * inv);
  }
}

// -------------------------------------- local window-5 attention (VALU)
__global__ __launch_bounds__(256) void local_attn_kernel(
    const _Float16* __restrict__ qkv, _Float16* __restrict__ out) {
  int idx = blockIdx.x * blockDim.x + threadIdx.x;
  if (idx >= TOK * 4) return;
  int h = idx & 3;
  int t = idx >> 2;
  int s = t & (SEQ - 1);
  const _Float16* q = qkv + (size_t)t * 768 + h * 64;

  float sc[5];
  const float scale = 0.125f;  // 1/sqrt(64)
#pragma unroll
  for (int j = 0; j < 5; ++j) {
    int ks = s + j - 2;
    if (ks < 0 || ks >= SEQ) { sc[j] = -1e30f; continue; }
    const _Float16* k = qkv + (size_t)(t + j - 2) * 768 + 256 + h * 64;
    float acc = 0.f;
    for (int d = 0; d < 64; ++d) acc += (float)q[d] * (float)k[d];
    sc[j] = acc * scale;
  }
  float mx = -1e30f;
#pragma unroll
  for (int j = 0; j < 5; ++j) mx = fmaxf(mx, sc[j]);
  float p[5], l = 0.f;
#pragma unroll
  for (int j = 0; j < 5; ++j) { p[j] = __expf(sc[j] - mx); l += p[j]; }
  float inv = fast_rcp(l);

  for (int d = 0; d < 64; ++d) {
    float acc = 0.f;
#pragma unroll
    for (int j = 0; j < 5; ++j) {
      int ks = s + j - 2;
      if (ks >= 0 && ks < SEQ)
        acc += p[j] * (float)qkv[(size_t)(t + j - 2) * 768 + 512 + h * 64 + d];
    }
    out[(size_t)t * DIM + h * 64 + d] = (_Float16)(acc * inv);
  }
}

// -------------------------------------- residual add + LayerNorm
__global__ __launch_bounds__(256) void add_ln_kernel(
    const float* __restrict__ xa, const float* __restrict__ xb,
    const float* __restrict__ g, const float* __restrict__ bt,
    float* __restrict__ outf, _Float16* __restrict__ outh) {
  const int row = blockIdx.x;
  const int tid = threadIdx.x;
  __shared__ float red[256];

  float v = xa[(size_t)row * DIM + tid] + xb[(size_t)row * DIM + tid];
  red[tid] = v;
  __syncthreads();
  for (int off = 128; off > 0; off >>= 1) {
    if (tid < off) red[tid] += red[tid + off];
    __syncthreads();
  }
  float mu = red[0] * (1.f / (float)DIM);
  __syncthreads();
  float d = v - mu;
  red[tid] = d * d;
  __syncthreads();
  for (int off = 128; off > 0; off >>= 1) {
    if (tid < off) red[tid] += red[tid + off];
    __syncthreads();
  }
  float var = red[0] * (1.f / (float)DIM);
  float y = d * rsqrtf(var + 1e-5f) * g[tid] + bt[tid];
  outf[(size_t)row * DIM + tid] = y;
  if (outh) outh[(size_t)row * DIM + tid] = (_Float16)y;
}

// ---------------------------------------------------------------- launcher
extern "C" void kernel_launch(void* const* d_in, const int* in_sizes, int n_in,
                              void* d_out, int out_size, void* d_ws,
                              size_t ws_size, hipStream_t stream) {
  const float* x      = (const float*)d_in[0];
  const float* g_in_w = (const float*)d_in[1];
  const float* g_in_b = (const float*)d_in[2];
  const float* g_out_w= (const float*)d_in[3];
  const float* g_out_b= (const float*)d_in[4];
  const float* t_in_w = (const float*)d_in[5];
  const float* t_in_b = (const float*)d_in[6];
  const float* t_out_w= (const float*)d_in[7];
  const float* t_out_b= (const float*)d_in[8];
  const float* fus_w1 = (const float*)d_in[9];
  const float* fus_b1 = (const float*)d_in[10];
  const float* fus_w2 = (const float*)d_in[11];
  const float* fus_b2 = (const float*)d_in[12];
  const float* ffn_w1 = (const float*)d_in[13];
  const float* ffn_b1 = (const float*)d_in[14];
  const float* ffn_w2 = (const float*)d_in[15];
  const float* ffn_b2 = (const float*)d_in[16];
  const float* gn_g   = (const float*)d_in[17];
  const float* gn_b   = (const float*)d_in[18];
  const float* fn_g   = (const float*)d_in[19];
  const float* fn_b   = (const float*)d_in[20];
  float* out = (float*)d_out;

  size_t off = 0;
  auto alloc = [&](size_t bytes) -> void* {
    off = (off + 255) & ~(size_t)255;
    void* p = (char*)d_ws + off;
    off += bytes;
    return p;
  };

  _Float16* x16    = (_Float16*)alloc((size_t)TOK * 256 * 2);
  _Float16* w_gin  = (_Float16*)alloc((size_t)768 * 256 * 2);
  _Float16* w_gout = (_Float16*)alloc((size_t)256 * 256 * 2);
  _Float16* w_tin  = (_Float16*)alloc((size_t)768 * 256 * 2);
  _Float16* w_tout = (_Float16*)alloc((size_t)256 * 256 * 2);
  _Float16* w_f1   = (_Float16*)alloc((size_t)512 * 512 * 2);
  _Float16* w_f2   = (_Float16*)alloc((size_t)256 * 512 * 2);
  _Float16* w_n1   = (_Float16*)alloc((size_t)512 * 256 * 2);
  _Float16* w_n2   = (_Float16*)alloc((size_t)256 * 512 * 2);
  _Float16* qkv_g  = (_Float16*)alloc((size_t)TOK * 768 * 2);
  _Float16* qkv_t  = (_Float16*)alloc((size_t)TOK * 768 * 2);
  _Float16* vt_g   = (_Float16*)alloc((size_t)BATCH * 8 * 32 * SEQ * 2);
  _Float16* o_g    = (_Float16*)alloc((size_t)TOK * 256 * 2);
  _Float16* o_t    = (_Float16*)alloc((size_t)TOK * 256 * 2);
  _Float16* cat16  = (_Float16*)alloc((size_t)TOK * 512 * 2);
  _Float16* h16    = (_Float16*)alloc((size_t)TOK * 512 * 2);
  float*    fusedf = (float*)alloc((size_t)TOK * 256 * 4);
  float*    x1f    = (float*)alloc((size_t)TOK * 256 * 4);
  _Float16* x1h    = (_Float16*)alloc((size_t)TOK * 256 * 2);
  float*    ffnf   = (float*)alloc((size_t)TOK * 256 * 4);

  auto cvt = [&](const float* src, _Float16* dst, int n) {
    cvt_f32_f16_kernel<<<(n + 255) / 256, 256, 0, stream>>>(src, dst, n);
  };
  cvt(x, x16, TOK * 256);
  cvt(g_in_w, w_gin, 768 * 256);
  cvt(g_out_w, w_gout, 256 * 256);
  cvt(t_in_w, w_tin, 768 * 256);
  cvt(t_out_w, w_tout, 256 * 256);
  cvt(fus_w1, w_f1, 512 * 512);
  cvt(fus_w2, w_f2, 256 * 512);
  cvt(ffn_w1, w_n1, 512 * 256);
  cvt(ffn_w2, w_n2, 256 * 512);

  // GEMM dispatch: h16-out (plain), h16-out (silu), f32-out (plain)
  auto gemm_h = [&](const _Float16* A, const _Float16* W, const float* bias,
                    _Float16* Ch, int M, int N, int K, int ldc) {
    dim3 grid(N / 128, M / 64);
    gemm_f16_kernel<0, false, true>
        <<<grid, 128, 0, stream>>>(A, W, bias, nullptr, Ch, K, ldc);
  };
  auto gemm_h_silu = [&](const _Float16* A, const _Float16* W,
                         const float* bias, _Float16* Ch, int M, int N, int K,
                         int ldc) {
    dim3 grid(N / 128, M / 64);
    gemm_f16_kernel<1, false, true>
        <<<grid, 128, 0, stream>>>(A, W, bias, nullptr, Ch, K, ldc);
  };
  auto gemm_f = [&](const _Float16* A, const _Float16* W, const float* bias,
                    float* Cf, int M, int N, int K, int ldc) {
    dim3 grid(N / 128, M / 64);
    gemm_f16_kernel<0, true, false>
        <<<grid, 128, 0, stream>>>(A, W, bias, Cf, nullptr, K, ldc);
  };

  // QKV projections (both paths)
  gemm_h(x16, w_gin, g_in_b, qkv_g, TOK, 768, 256, 768);
  gemm_h(x16, w_tin, t_in_b, qkv_t, TOK, 768, 256, 768);

  // Global attention
  transpose_vg_kernel<<<TOK, 256, 0, stream>>>(qkv_g, vt_g);
  flash_global_kernel<<<dim3(SEQ / 64, 8, BATCH), 128, 0, stream>>>(qkv_g, vt_g,
                                                                    o_g);
  // Local attention
  local_attn_kernel<<<(TOK * 4 + 255) / 256, 256, 0, stream>>>(qkv_t, o_t);

  // Out-projections write directly into concat buffer [8192, 512]
  gemm_h(o_g, w_gout, g_out_b, cat16, TOK, 256, 256, 512);
  gemm_h(o_t, w_tout, t_out_b, cat16 + 256, TOK, 256, 256, 512);

  // Fusion MLP
  gemm_h_silu(cat16, w_f1, fus_b1, h16, TOK, 512, 512, 512);
  gemm_f(h16, w_f2, fus_b2, fusedf, TOK, 256, 512, 256);
  add_ln_kernel<<<TOK, 256, 0, stream>>>(x, fusedf, gn_g, gn_b, x1f, x1h);

  // FFN
  gemm_h_silu(x1h, w_n1, ffn_b1, h16, TOK, 512, 256, 512);
  gemm_f(h16, w_n2, ffn_b2, ffnf, TOK, 256, 512, 256);
  add_ln_kernel<<<TOK, 256, 0, stream>>>(x1f, ffnf, fn_g, fn_b, out, nullptr);
}
